// GNN_1_WithWater_75986561401175
// MI455X (gfx1250) — compile-verified, matrix operand order checked by reading, not run on hardware
//
#include <hip/hip_runtime.h>
#include <hip/hip_bf16.h>

// ---------------------------------------------------------------------------
// Types for CDNA5 WMMA (wave32)
// ---------------------------------------------------------------------------
typedef __bf16 v16bf __attribute__((ext_vector_type(16)));
typedef float  v8f   __attribute__((ext_vector_type(8)));

#define EMB    128
#define NATOM  133
#define KPAD   160        // NATOM padded up to multiple of 32
#define NLAYER 3
#define BN_EPS 1e-5f

__device__ __forceinline__ void atomic_add_f32(float* p, float v) {
    __hip_atomic_fetch_add(p, v, __ATOMIC_RELAXED, __HIP_MEMORY_SCOPE_AGENT);
}

// A-fragment (16x32 bf16, MxK) gather from an LDS tile laid out row-major
// [16 rows][ldm cols].  Per ISA 7.12.2: lanes 0-15 hold M=lane with
// K={0..7,16..23}; lanes 16-31 hold M=lane-16 with K={8..15,24..31}.
// Each 8-element group is contiguous -> compiler emits 2x ds_load_b128.
__device__ __forceinline__ v16bf load_afrag_lds(const __bf16* tile, int ldm,
                                                int lane, int kbase) {
    v16bf a;
    const int m     = lane & 15;
    const int half8 = ((lane >> 4) & 1) << 3;
    const __bf16* rowp = tile + m * ldm + kbase;
#pragma unroll
    for (int i = 0; i < 16; ++i) {
        const int k = ((i >> 3) << 4) + half8 + (i & 7);
        a[i] = rowp[k];
    }
    return a;
}

// B-fragment from pre-swizzled (fragment-ordered) weights: one contiguous
// 32-byte aligned vector load per lane.
__device__ __forceinline__ v16bf load_bfrag_sw(const __bf16* wsw, int frag,
                                               int lane) {
    return *(const v16bf*)(wsw + (((size_t)frag * 32 + lane) << 4));
}

__device__ __forceinline__ v8f wmma_bf16(v16bf a, v16bf b, v8f c) {
    return __builtin_amdgcn_wmma_f32_16x16x32_bf16(
        /*neg_a=*/false, a, /*neg_b=*/false, b,
        /*c_mod=*/(short)0, c, /*reuse_a=*/false, /*reuse_b=*/false);
}

// ---------------------------------------------------------------------------
// Weight swizzle: row-major fp32 [Ksrc..Kp, N] -> bf16 fragment order.
//   dst[((ktile*(N/16) + ntile)*32 + lane)*16 + i] = w[k(i,lane), n(lane)]
// Rows k >= Ksrc are zero (pads NATOM=133 up to KPAD=160).
// ---------------------------------------------------------------------------
__global__ void swz_w_kernel(__bf16* __restrict__ dst, const float* __restrict__ src,
                             int Kp, int N, int Ksrc) {
    const int idx = blockIdx.x * 256 + threadIdx.x;
    if (idx >= Kp * N) return;
    const int i      = idx & 15;
    const int lane   = (idx >> 4) & 31;
    const int t      = idx >> 9;
    const int ntiles = N >> 4;
    const int ntile  = t % ntiles;
    const int ktile  = t / ntiles;
    const int half8  = ((lane >> 4) & 1) << 3;
    const int n = ntile * 16 + (lane & 15);
    const int k = ktile * 32 + ((i >> 3) << 4) + half8 + (i & 7);
    const float v = (k < Ksrc) ? src[(size_t)k * N + n] : 0.f;
    dst[idx] = (__bf16)v;
}

// ---------------------------------------------------------------------------
// h0 = x[N,133] @ Wx + bx  (one wave per 16-row node tile)
// ---------------------------------------------------------------------------
__global__ __launch_bounds__(32)
void proj_kernel(const float* __restrict__ x, const __bf16* __restrict__ wx_sw,
                 const float* __restrict__ bx, float* __restrict__ h_out, int N) {
    __shared__ __bf16 tileA[16 * KPAD];
    const int lane = threadIdx.x;
    const int row0 = blockIdx.x * 16;
    const int rows = (N - row0) < 16 ? (N - row0) : 16;

    for (int idx = lane; idx < 16 * KPAD; idx += 32) {
        const int r = idx / KPAD;
        const int c = idx - r * KPAD;
        float v = 0.f;
        if (r < rows && c < NATOM) v = x[(size_t)(row0 + r) * NATOM + c];
        tileA[idx] = (__bf16)v;
    }
    __syncthreads();

    v16bf afr[5];
#pragma unroll
    for (int kk = 0; kk < 5; ++kk) afr[kk] = load_afrag_lds(tileA, KPAD, lane, kk * 32);

    const int n15   = lane & 15;
    const int mbase = ((lane >> 4) & 1) << 3;
#pragma unroll
    for (int nt = 0; nt < 8; ++nt) {
        v8f acc = {};
#pragma unroll
        for (int kk = 0; kk < 5; ++kk)
            acc = wmma_bf16(afr[kk], load_bfrag_sw(wx_sw, kk * 8 + nt, lane), acc);
        const int ncol = nt * 16 + n15;
        const float bias = bx[ncol];
#pragma unroll
        for (int r = 0; r < 8; ++r) {
            const int row = mbase + r;
            if (row < rows)
                h_out[(size_t)(row0 + row) * EMB + ncol] = acc[r] + bias;
        }
    }
}

// ---------------------------------------------------------------------------
// agg[dst] += h[src] : one 32-lane group per edge, float4 per lane
// ---------------------------------------------------------------------------
__global__ __launch_bounds__(256)
void scatter_kernel(const float* __restrict__ h, const int* __restrict__ ei,
                    float* __restrict__ agg, int E) {
    const int e = blockIdx.x * 8 + (threadIdx.x >> 5);
    if (e >= E) return;
    const int lane = threadIdx.x & 31;
    const int src = ei[e];
    const int dst = ei[E + e];
    const float4 v = *(const float4*)(h + (size_t)src * EMB + lane * 4);
    float* p = agg + (size_t)dst * EMB + lane * 4;
    atomic_add_f32(p + 0, v.x);
    atomic_add_f32(p + 1, v.y);
    atomic_add_f32(p + 2, v.z);
    atomic_add_f32(p + 3, v.w);
}

// ---------------------------------------------------------------------------
// Per-layer GIN MLP, fused residual + BN-stat accumulation.
//   m = relu((h+agg) @ w1 + b1) @ w2 + b2 ; h_out = m + h
//   bn_sum[f] += sum_rows h_out ; bn_sum[128+f] += sum_rows h_out^2
// One wave per 16-row node tile; weights pre-swizzled to fragment order.
// ---------------------------------------------------------------------------
__global__ __launch_bounds__(32)
void mlp_kernel(const float* __restrict__ h, const float* __restrict__ agg,
                const __bf16* __restrict__ w1_sw, const float* __restrict__ b1,
                const __bf16* __restrict__ w2_sw, const float* __restrict__ b2,
                float* __restrict__ h_out, float* __restrict__ bn_sum, int N) {
    __shared__ __bf16 tileA[16 * EMB];        // (h+agg) tile, bf16
    __shared__ __bf16 inter[16 * 2 * EMB];    // relu(.@w1+b1), bf16
    const int lane = threadIdx.x;
    const int row0 = blockIdx.x * 16;
    const int rows = (N - row0) < 16 ? (N - row0) : 16;

    // stage (h + agg) -> LDS bf16, vectorized
    for (int t = lane; t < 16 * EMB / 4; t += 32) {
        const int idx = t * 4;
        const int r = idx >> 7, c = idx & 127;
        float4 hv = make_float4(0.f, 0.f, 0.f, 0.f), av = hv;
        if (r < rows) {
            hv = *(const float4*)(h   + (size_t)(row0 + r) * EMB + c);
            av = *(const float4*)(agg + (size_t)(row0 + r) * EMB + c);
        }
        tileA[idx + 0] = (__bf16)(hv.x + av.x);
        tileA[idx + 1] = (__bf16)(hv.y + av.y);
        tileA[idx + 2] = (__bf16)(hv.z + av.z);
        tileA[idx + 3] = (__bf16)(hv.w + av.w);
    }
    __syncthreads();

    const int n15   = lane & 15;
    const int mbase = ((lane >> 4) & 1) << 3;

    // GEMM1: [16,128] x [128,256]  (16 n-tiles, K in 4 steps of 32)
    v16bf afr[4];
#pragma unroll
    for (int kk = 0; kk < 4; ++kk) afr[kk] = load_afrag_lds(tileA, EMB, lane, kk * 32);
#pragma unroll
    for (int nt = 0; nt < 16; ++nt) {
        // prefetch next n-tile's weight fragments into L2/L0
        __builtin_prefetch(w1_sw + (((size_t)(nt + 1) * 32 + lane) << 4), 0, 1);
        v8f acc = {};
#pragma unroll
        for (int kk = 0; kk < 4; ++kk)
            acc = wmma_bf16(afr[kk], load_bfrag_sw(w1_sw, kk * 16 + nt, lane), acc);
        const int ncol = nt * 16 + n15;
        const float bias = b1[ncol];
#pragma unroll
        for (int r = 0; r < 8; ++r) {
            float v = acc[r] + bias;
            v = fmaxf(v, 0.f);
            inter[(mbase + r) * (2 * EMB) + ncol] = (__bf16)v;
        }
    }
    __syncthreads();

    // GEMM2: [16,256] x [256,128], fused residual + BN stats
#pragma unroll
    for (int nt = 0; nt < 8; ++nt) {
        __builtin_prefetch(w2_sw + (((size_t)(nt + 1) * 32 + lane) << 4), 0, 1);
        v8f acc = {};
#pragma unroll
        for (int kk = 0; kk < 8; ++kk)
            acc = wmma_bf16(load_afrag_lds(inter, 2 * EMB, lane, kk * 32),
                            load_bfrag_sw(w2_sw, kk * 8 + nt, lane), acc);
        const int ncol = nt * 16 + n15;
        const float bias = b2[ncol];
        float s = 0.f, s2 = 0.f;
#pragma unroll
        for (int r = 0; r < 8; ++r) {
            const int row = mbase + r;
            if (row < rows) {
                const size_t off = (size_t)(row0 + row) * EMB + ncol;
                const float v = acc[r] + bias + h[off];   // residual
                h_out[off] = v;
                s += v;
                s2 += v * v;
            }
        }
        atomic_add_f32(&bn_sum[ncol], s);
        atomic_add_f32(&bn_sum[EMB + ncol], s2);
    }
}

// ---------------------------------------------------------------------------
// BN finalize + apply (+ optional relu), in place
// ---------------------------------------------------------------------------
__global__ void bn_kernel(float* __restrict__ h, const float* __restrict__ bn,
                          const float* __restrict__ g, const float* __restrict__ be,
                          float invN, int total, int do_relu) {
    const int i = blockIdx.x * 256 + threadIdx.x;
    if (i >= total) return;
    const int f = i & 127;
    const float mu  = bn[f] * invN;
    const float var = bn[EMB + f] * invN - mu * mu;
    float v = (h[i] - mu) * rsqrtf(var + BN_EPS) * g[f] + be[f];
    if (do_relu) v = fmaxf(v, 0.f);
    h[i] = v;
}

// ---------------------------------------------------------------------------
// Weighted mean-pool accumulation (mask == nullptr -> weight 1)
// ---------------------------------------------------------------------------
__global__ __launch_bounds__(256)
void pool_kernel(const float* __restrict__ h, const int* __restrict__ batch,
                 const int* __restrict__ mask, float* __restrict__ pool,
                 float* __restrict__ cnt, int N) {
    const int node = blockIdx.x * 8 + (threadIdx.x >> 5);
    if (node >= N) return;
    if (mask && mask[node] <= 0) return;
    const int lane = threadIdx.x & 31;
    const int b = batch[node];
    const float4 v = *(const float4*)(h + (size_t)node * EMB + lane * 4);
    float* p = pool + (size_t)b * EMB + lane * 4;
    atomic_add_f32(p + 0, v.x);
    atomic_add_f32(p + 1, v.y);
    atomic_add_f32(p + 2, v.z);
    atomic_add_f32(p + 3, v.w);
    if (lane == 0) atomic_add_f32(&cnt[b], 1.f);
}

// ---------------------------------------------------------------------------
// FC head: rep[256] -> 128 -> 64 -> 1, one block per graph
// ---------------------------------------------------------------------------
__global__ __launch_bounds__(128)
void fc_kernel(const float* __restrict__ pool_s, const float* __restrict__ cnt_s,
               const float* __restrict__ pool_h, const float* __restrict__ cnt_h,
               const float* __restrict__ fcw1, const float* __restrict__ fcb1,
               const float* __restrict__ fcw2, const float* __restrict__ fcb2,
               const float* __restrict__ fcw3, const float* __restrict__ fcb3,
               float* __restrict__ out) {
    __shared__ float rep[256];
    __shared__ float h1[128];
    __shared__ float h2[64];
    const int g = blockIdx.x, t = threadIdx.x;
    const float cs = fmaxf(cnt_s[g], 1.f);
    const float ch = fmaxf(cnt_h[g], 1.f);
    rep[t]       = pool_s[(size_t)g * EMB + t] / cs;
    rep[128 + t] = pool_h[(size_t)g * EMB + t] / ch;
    __syncthreads();
    float a = fcb1[t];
    for (int k = 0; k < 256; ++k) a += rep[k] * fcw1[k * 128 + t];
    h1[t] = fmaxf(a, 0.f);
    __syncthreads();
    if (t < 64) {
        float a2 = fcb2[t];
        for (int k = 0; k < 128; ++k) a2 += h1[k] * fcw2[k * 64 + t];
        h2[t] = fmaxf(a2, 0.f);
    }
    __syncthreads();
    if (t == 0) {
        float a3 = fcb3[0];
        for (int k = 0; k < 64; ++k) a3 += h2[k] * fcw3[k];
        out[g] = a3;
    }
}

// ---------------------------------------------------------------------------
// Host orchestration
// ---------------------------------------------------------------------------
extern "C" void kernel_launch(void* const* d_in, const int* in_sizes, int n_in,
                              void* d_out, int out_size, void* d_ws, size_t ws_size,
                              hipStream_t stream) {
    const float* solute_x = (const float*)d_in[0];
    const float* hyd_x    = (const float*)d_in[1];
    const float* Wx_s = (const float*)d_in[2];
    const float* bx_s = (const float*)d_in[3];
    const float* Wx_h = (const float*)d_in[4];
    const float* bx_h = (const float*)d_in[5];
    const float* w1_s = (const float*)d_in[6];
    const float* b1_s = (const float*)d_in[7];
    const float* w2_s = (const float*)d_in[8];
    const float* b2_s = (const float*)d_in[9];
    const float* g_s  = (const float*)d_in[10];
    const float* be_s = (const float*)d_in[11];
    const float* w1_h = (const float*)d_in[12];
    const float* b1_h = (const float*)d_in[13];
    const float* w2_h = (const float*)d_in[14];
    const float* b2_h = (const float*)d_in[15];
    const float* g_h  = (const float*)d_in[16];
    const float* be_h = (const float*)d_in[17];
    const float* fcw1 = (const float*)d_in[18];
    const float* fcb1 = (const float*)d_in[19];
    const float* fcw2 = (const float*)d_in[20];
    const float* fcb2 = (const float*)d_in[21];
    const float* fcw3 = (const float*)d_in[22];
    const float* fcb3 = (const float*)d_in[23];
    const int* ei_s    = (const int*)d_in[24];
    const int* ei_h    = (const int*)d_in[25];
    const int* batch_s = (const int*)d_in[26];
    const int* batch_h = (const int*)d_in[27];
    const int* mask_h  = (const int*)d_in[28];

    const int NS = in_sizes[0] / NATOM;
    const int NH = in_sizes[1] / NATOM;
    const int ES = in_sizes[24] / 2;
    const int EH = in_sizes[25] / 2;
    const int NG = out_size;          // num_tasks == 1

    // ----- workspace carve-out (branches processed sequentially, reuse) -----
    char* ws = (char*)d_ws;
    size_t off = 0;
    auto wsalloc = [&](size_t bytes) -> void* {
        void* p = ws + off;
        off = (off + bytes + 255) & ~(size_t)255;
        return p;
    };
    const size_t NMAX = (size_t)(NS > NH ? NS : NH);
    const size_t W1SZ = (size_t)EMB * 2 * EMB;     // per-layer elements
    const size_t W2SZ = (size_t)2 * EMB * EMB;
    float*  hA     = (float*)wsalloc(NMAX * EMB * 4);
    float*  hB     = (float*)wsalloc(NMAX * EMB * 4);
    float*  agg    = (float*)wsalloc(NMAX * EMB * 4);
    __bf16* wx_sw  = (__bf16*)wsalloc((size_t)KPAD * EMB * 2);
    __bf16* w1_sw  = (__bf16*)wsalloc((size_t)NLAYER * W1SZ * 2);
    __bf16* w2_sw  = (__bf16*)wsalloc((size_t)NLAYER * W2SZ * 2);
    float*  bn     = (float*)wsalloc(2 * EMB * 4);
    float*  pool_s = (float*)wsalloc((size_t)NG * EMB * 4);
    float*  cnt_s  = (float*)wsalloc((size_t)NG * 4);
    float*  pool_h = (float*)wsalloc((size_t)NG * EMB * 4);
    float*  cnt_h  = (float*)wsalloc((size_t)NG * 4);
    (void)ws_size; (void)n_in;

    auto cdiv = [](int a, int b) { return (a + b - 1) / b; };

    auto run_branch = [&](const float* x, const float* Wx, const float* bx,
                          const float* w1, const float* b1,
                          const float* w2, const float* b2,
                          const float* g, const float* be,
                          const int* ei, int E, int N,
                          const int* batch, const int* mask,
                          float* pool, float* cnt) {
        // swizzle weights into bf16 fragment order (per layer)
        swz_w_kernel<<<cdiv(KPAD * EMB, 256), 256, 0, stream>>>(
            wx_sw, Wx, KPAD, EMB, NATOM);
        for (int l = 0; l < NLAYER; ++l) {
            swz_w_kernel<<<cdiv((int)W1SZ, 256), 256, 0, stream>>>(
                w1_sw + l * W1SZ, w1 + l * W1SZ, EMB, 2 * EMB, EMB);
            swz_w_kernel<<<cdiv((int)W2SZ, 256), 256, 0, stream>>>(
                w2_sw + l * W2SZ, w2 + l * W2SZ, 2 * EMB, EMB, 2 * EMB);
        }

        // h0 = x @ Wx + bx
        proj_kernel<<<cdiv(N, 16), 32, 0, stream>>>(x, wx_sw, bx, hA, N);

        float* hcur = hA;
        float* hnxt = hB;
        for (int l = 0; l < NLAYER; ++l) {
            hipMemsetAsync(agg, 0, (size_t)N * EMB * 4, stream);
            scatter_kernel<<<cdiv(E, 8), 256, 0, stream>>>(hcur, ei, agg, E);
            hipMemsetAsync(bn, 0, 2 * EMB * 4, stream);
            mlp_kernel<<<cdiv(N, 16), 32, 0, stream>>>(
                hcur, agg,
                w1_sw + (size_t)l * W1SZ, b1 + (size_t)l * 2 * EMB,
                w2_sw + (size_t)l * W2SZ, b2 + (size_t)l * EMB,
                hnxt, bn, N);
            bn_kernel<<<cdiv(N * EMB, 256), 256, 0, stream>>>(
                hnxt, bn, g + (size_t)l * EMB, be + (size_t)l * EMB,
                1.f / (float)N, N * EMB, (l < NLAYER - 1) ? 1 : 0);
            float* t = hcur; hcur = hnxt; hnxt = t;
        }
        hipMemsetAsync(pool, 0, (size_t)NG * EMB * 4, stream);
        hipMemsetAsync(cnt, 0, (size_t)NG * 4, stream);
        pool_kernel<<<cdiv(N, 8), 256, 0, stream>>>(hcur, batch, mask, pool, cnt, N);
    };

    run_branch(solute_x, Wx_s, bx_s, w1_s, b1_s, w2_s, b2_s, g_s, be_s,
               ei_s, ES, NS, batch_s, nullptr, pool_s, cnt_s);
    run_branch(hyd_x, Wx_h, bx_h, w1_h, b1_h, w2_h, b2_h, g_h, be_h,
               ei_h, EH, NH, batch_h, mask_h, pool_h, cnt_h);

    fc_kernel<<<NG, 128, 0, stream>>>(pool_s, cnt_s, pool_h, cnt_h,
                                      fcw1, fcb1, fcw2, fcb2, fcw3, fcb3,
                                      (float*)d_out);
}
